// SeqSelfAttention_68874095558761
// MI455X (gfx1250) — matrix operand order repaired and checked by reading, not compiled, not run on hardware
//
#include <hip/hip_runtime.h>
#include <hip/hip_bf16.h>

typedef __attribute__((ext_vector_type(16))) _Float16 v16h;
typedef __attribute__((ext_vector_type(8)))  _Float16 v8h;
typedef __attribute__((ext_vector_type(4)))  _Float16 v4h;
typedef __attribute__((ext_vector_type(8)))  float    v8f;

#define B_   8
#define L_   512
#define D_   128
#define U_   32
#define TM   32          // query rows per block
#define KS   160         // banded key strip per block: [t0-64, t0+96)
#define KSP  168         // padded LDS row stride (halves): 336B, 16B-aligned rows
#define EPS  1e-7f

// One fused kernel: block = (b, t-tile of 32 queries), 256 threads = 8 wave32.
__global__ __launch_bounds__(256)
void seqattn_kernel(const float* __restrict__ x,  const float* __restrict__ Wt,
                    const float* __restrict__ Wx, const float* __restrict__ bh,
                    const float* __restrict__ Wa, const float* __restrict__ ba,
                    float* __restrict__ out) {
    __shared__ __align__(16) _Float16 As[TM][KSP];   // (a_s - w_out)*inv, banded
    __shared__ __align__(16) _Float16 XsT[D_][KSP];  // fp16 value strip, TRANSPOSED [d][k]
    __shared__ float sWtv[D_], sWxv[D_];
    __shared__ float sEs[KS], sEt[TM], sCol[D_], sWsc[TM];
    __shared__ float sC[1];

    const int tid  = threadIdx.x;
    const int lane = tid & 31;
    const int w    = tid >> 5;           // wave id 0..7 (wave32!)
    const int b    = blockIdx.y;
    const int t0   = blockIdx.x * TM;
    const int s0   = t0 - 64;            // strip start (may be negative)

    // ---- Phase A: projected score vectors wtv = Wt@Wa, wxv = Wx@Wa, c ----
    if (tid < D_) {
        float s = 0.f;
        #pragma unroll
        for (int u = 0; u < U_; ++u) s += Wt[tid * U_ + u] * Wa[u];
        sWtv[tid] = s;
    } else {
        int d = tid - D_;
        float s = 0.f;
        #pragma unroll
        for (int u = 0; u < U_; ++u) s += Wx[d * U_ + u] * Wa[u];
        sWxv[d] = s;
    }
    if (tid == 0) {
        float s = 0.f;
        #pragma unroll
        for (int u = 0; u < U_; ++u) s += bh[u] * Wa[u];
        sC[0] = s + ba[0];
    }

    // ---- Phase A2: stage fp16 value strip TRANSPOSED into LDS ----
    // Thread (d, half) owns column d, k-range [80*half, 80*half+80).
    // Global reads coalesced across threads (consecutive d at same s);
    // LDS writes contiguous per thread -> ds_store_b64.
    {
        const int d  = tid & (D_ - 1);
        const int hf = tid >> 7;               // 0 or 1
        const int k0 = hf * 80;
        for (int ls = k0; ls < k0 + 80; ls += 4) {
            v4h h;
            #pragma unroll
            for (int q = 0; q < 4; ++q) {
                int s = s0 + ls + q;
                float v = (s >= 0 && s < L_) ? x[((size_t)b * L_ + s) * D_ + d] : 0.f;
                h[q] = (_Float16)v;
            }
            *reinterpret_cast<v4h*>(&XsT[d][ls]) = h;   // 8B-aligned: ls%4==0, row 336B
        }
    }
    __syncthreads();

    // ---- Phase B: es strip, et rows, column sums of x[b] ----
    if (tid < KS) {
        int s = s0 + tid;
        float v = 0.f;
        if (s >= 0 && s < L_) {
            const float* xr = x + ((size_t)b * L_ + s) * D_;
            #pragma unroll 8
            for (int d = 0; d < D_; ++d) v += xr[d] * sWxv[d];
        }
        sEs[tid] = v;
    } else if (tid < KS + TM) {
        int r = tid - KS;
        const float* xr = x + ((size_t)b * L_ + (t0 + r)) * D_;
        float v = 0.f;
        #pragma unroll 8
        for (int d = 0; d < D_; ++d) v += xr[d] * sWtv[d];
        sEt[r] = v;
    } else {
        int c0 = (tid - (KS + TM)) * 2;      // 64 threads x 2 cols
        float a0 = 0.f, a1 = 0.f;
        const float* xb = x + (size_t)b * L_ * D_;
        for (int s = 0; s < L_; ++s) {
            a0 += xb[s * D_ + c0];
            a1 += xb[s * D_ + c0 + 1];
        }
        sCol[c0] = a0; sCol[c0 + 1] = a1;
    }
    __syncthreads();

    // ---- Phase D: fp32 softmax over banded scores (zeros elsewhere) ----
    const float cc = sC[0];
    #pragma unroll
    for (int rr = 0; rr < 4; ++rr) {
        const int tl = w * 4 + rr;           // local row 0..31
        const int t  = t0 + tl;
        const int lo = max(t - 63, 0);
        const int hi = min(t + 65, L_);      // exclusive
        const int cnt = hi - lo;
        const float etv = sEt[tl];
        float ev[4];
        float mx = 0.f;                      // masked zeros always participate
        #pragma unroll
        for (int i = 0; i < 4; ++i) {
            int idx = lo + lane + 32 * i;
            float e = 0.f;
            if (idx < hi) { e = etv + sEs[idx - s0] + cc; mx = fmaxf(mx, e); }
            ev[i] = e;
        }
        #pragma unroll
        for (int off = 16; off >= 1; off >>= 1)
            mx = fmaxf(mx, __shfl_xor(mx, off, 32));
        float pv[4], sum = 0.f;
        #pragma unroll
        for (int i = 0; i < 4; ++i) {
            int idx = lo + lane + 32 * i;
            float p = 0.f;
            if (idx < hi) { p = __expf(ev[i] - mx); sum += p; }
            pv[i] = p;
        }
        #pragma unroll
        for (int off = 16; off >= 1; off >>= 1)
            sum += __shfl_xor(sum, off, 32);
        const float wout  = __expf(-mx);
        const float denom = sum + (float)(L_ - cnt) * wout + EPS;
        const float inv   = 1.f / denom;
        #pragma unroll
        for (int i = 0; i < 5; ++i) As[tl][lane + 32 * i] = (_Float16)0.f;
        #pragma unroll
        for (int i = 0; i < 4; ++i) {
            int idx = lo + lane + 32 * i;
            if (idx < hi) As[tl][idx - s0] = (_Float16)((pv[i] - wout) * inv);
        }
        if (lane == 0) sWsc[tl] = wout * inv;
    }
    __syncthreads();

    // ---- Phase F: banded A(32x160,f16) @ X(160x128,f16) via WMMA ----
    const int g   = lane >> 4;               // lane half
    const int ln  = lane & 15;
    const int m   = (w & 1) * 16;            // M strip
    const int nb0 = (w >> 1) * 32;           // two N tiles: nb0, nb0+16
    v8f acc0 = {}; v8f acc1 = {};
    #pragma unroll
    for (int kb = 0; kb < KS; kb += 32) {
        // A layout (contiguous, matching B): lane-half g holds
        // K = kb + 16g + {0..15}
        v8h a_lo = *reinterpret_cast<const v8h*>(&As[m + ln][kb + (g << 4)]);
        v8h a_hi = *reinterpret_cast<const v8h*>(&As[m + ln][kb + (g << 4) + 8]);
        v16h af  = __builtin_shufflevector(a_lo, a_hi,
                       0,1,2,3,4,5,6,7,8,9,10,11,12,13,14,15);
        // B layout (dense 16-bit B 32x16): lane-half g holds K = kb + 16g + j.
        // XsT is [n][k] so the per-lane 16-K span is contiguous -> 2x ds_load_b128.
        v8h b0l = *reinterpret_cast<const v8h*>(&XsT[nb0 + ln][kb + (g << 4)]);
        v8h b0h = *reinterpret_cast<const v8h*>(&XsT[nb0 + ln][kb + (g << 4) + 8]);
        v16h bf0 = __builtin_shufflevector(b0l, b0h,
                       0,1,2,3,4,5,6,7,8,9,10,11,12,13,14,15);
        v8h b1l = *reinterpret_cast<const v8h*>(&XsT[nb0 + 16 + ln][kb + (g << 4)]);
        v8h b1h = *reinterpret_cast<const v8h*>(&XsT[nb0 + 16 + ln][kb + (g << 4) + 8]);
        v16h bf1 = __builtin_shufflevector(b1l, b1h,
                       0,1,2,3,4,5,6,7,8,9,10,11,12,13,14,15);

        acc0 = __builtin_amdgcn_wmma_f32_16x16x32_f16(
                   false, af, false, bf0, (short)0, acc0, false, false);
        acc1 = __builtin_amdgcn_wmma_f32_16x16x32_f16(
                   false, af, false, bf1, (short)0, acc1, false, false);
    }

    // ---- Phase G: rank-1 correction w_out*colsum, store fp32 ----
    #pragma unroll
    for (int r = 0; r < 8; ++r) {
        int row = m + r + (g << 3);          // C/D layout: M = r + 8*laneHalf
        int t   = t0 + row;
        float ws = sWsc[row];
        size_t base = ((size_t)b * L_ + t) * D_;
        out[base + nb0 + ln]      = acc0[r] + ws * sCol[nb0 + ln];
        out[base + nb0 + 16 + ln] = acc1[r] + ws * sCol[nb0 + 16 + ln];
    }
}

extern "C" void kernel_launch(void* const* d_in, const int* in_sizes, int n_in,
                              void* d_out, int out_size, void* d_ws, size_t ws_size,
                              hipStream_t stream) {
    (void)in_sizes; (void)n_in; (void)d_ws; (void)ws_size; (void)out_size;
    const float* x  = (const float*)d_in[0];
    const float* Wt = (const float*)d_in[1];
    const float* Wx = (const float*)d_in[2];
    const float* bh = (const float*)d_in[3];
    const float* Wa = (const float*)d_in[4];
    const float* ba = (const float*)d_in[5];
    float* out = (float*)d_out;
    dim3 grid(L_ / TM, B_);   // 16 x 8 = 128 blocks
    seqattn_kernel<<<grid, 256, 0, stream>>>(x, Wt, Wx, bh, Wa, ba, out);
}
